// MultiHeadAttention_torch_25271587569876
// MI455X (gfx1250) — matrix operand order repaired and checked by reading, not compiled, not run on hardware
//
#include <hip/hip_runtime.h>

// ---------------------------------------------------------------------------
// MultiHeadAttention (B=2, S=2048, D=1024, H=16, HD=64) for gfx1250 (MI455X)
// f16 WMMA (v_wmma_f32_16x16x32_f16) with f32 accumulation, flash-attention.
// GEMMs: 32x64 tile per wave, ping-pong double-buffered over K (no reg copies).
// Attention: K fragments preloaded one iteration ahead, V hoisted over softmax.
// ---------------------------------------------------------------------------

#define B_  2
#define S_  2048
#define D_  1024
#define H_  16
#define HD_ 64

typedef __attribute__((ext_vector_type(16))) _Float16 v16h;
typedef __attribute__((ext_vector_type(8)))  _Float16 v8h;
typedef __attribute__((ext_vector_type(4)))  _Float16 v4h;
typedef __attribute__((ext_vector_type(8)))  float    v8f;

#define WMMA_F32_F16(a, b, c) \
  __builtin_amdgcn_wmma_f32_16x16x32_f16(false, (a), false, (b), (short)0, (c), false, false)

// A fragment: 16x32 (MxK) f16, row-major source, row stride ld (elements).
// lane<16  holds row (lane)   : K {0..15}
// lane>=16 holds row (lane-16): K {16..31}
__device__ __forceinline__ v16h load_a_frag(const _Float16* base, int ld) {
  const int lane = threadIdx.x & 31;
  const int m    = lane & 15;
  const int off  = (lane >> 4) * 16;
  const _Float16* p = base + m * ld + off;
  union { v16h v; v8h h[2]; } u;
  u.h[0] = *(const v8h*)(p);       // K = off .. off+7
  u.h[1] = *(const v8h*)(p + 8);   // K = off+8 .. off+15
  return u.v;
}

// B fragment: 32x16 (KxN) f16. lane n (mod 16) holds column n, K contiguous.
// Source: pointer to (n=0,k=0); lane reads 16 contiguous elements of source
// row n (row stride ld), starting at K = 16*(lane>>4).
__device__ __forceinline__ v16h load_b_frag(const _Float16* base, int ld) {
  const int lane = threadIdx.x & 31;
  const int n    = lane & 15;
  const int koff = (lane >> 4) * 16;
  const _Float16* p = base + n * ld + koff;
  union { v16h v; v8h h[2]; } u;
  u.h[0] = *(const v8h*)(p);       // K = koff .. koff+7
  u.h[1] = *(const v8h*)(p + 8);   // K = koff+8 .. koff+15
  return u.v;
}

// ---------------------------------------------------------------------------
// f32 -> f16 conversion (4-wide)
// ---------------------------------------------------------------------------
__global__ __launch_bounds__(256)
void f32_to_f16_kernel(const float* __restrict__ src, _Float16* __restrict__ dst, int n4) {
  int i = blockIdx.x * blockDim.x + threadIdx.x;
  if (i < n4) {
    const float4 f = ((const float4*)src)[i];
    v4h o = { (_Float16)f.x, (_Float16)f.y, (_Float16)f.z, (_Float16)f.w };
    ((v4h*)dst)[i] = o;
  }
}

// ---------------------------------------------------------------------------
// QKV projection: Y = X @ W^T + b   (M=B*S=4096, N=D, K=D), f16 in, f16 out.
// mode 0/1 -> out laid out [B,H,S,HD]; mode 2 -> out laid out [B,H,HD,S].
// 4 waves per block; each wave computes a 32x64 tile; ping-pong over K.
// ---------------------------------------------------------------------------
__global__ __launch_bounds__(128)
void proj_kernel(const _Float16* __restrict__ Xh, const _Float16* __restrict__ Wh,
                 const float* __restrict__ bias, _Float16* __restrict__ out, int mode) {
  const int wave  = threadIdx.x >> 5;
  const int mBase = (blockIdx.x * 4 + wave) * 32;
  const int nBase = blockIdx.y * 64;

  v8f acc[2][4] = {};

  // set 0: fragments for k = 0
  v16h a0 = load_a_frag(Xh + mBase * D_, D_);
  v16h a1 = load_a_frag(Xh + (mBase + 16) * D_, D_);
  v16h b0[4];
#pragma unroll
  for (int t = 0; t < 4; ++t) b0[t] = load_b_frag(Wh + (nBase + t * 16) * D_, D_);

#pragma unroll 1
  for (int k = 0; k < D_; k += 64) {
    // set 1 loads (k+32), issued before set-0 WMMAs
    v16h c0 = load_a_frag(Xh + mBase * D_ + k + 32, D_);
    v16h c1 = load_a_frag(Xh + (mBase + 16) * D_ + k + 32, D_);
    v16h b1[4];
#pragma unroll
    for (int t = 0; t < 4; ++t) b1[t] = load_b_frag(Wh + (nBase + t * 16) * D_ + k + 32, D_);

#pragma unroll
    for (int t = 0; t < 4; ++t) {
      acc[0][t] = WMMA_F32_F16(a0, b0[t], acc[0][t]);
      acc[1][t] = WMMA_F32_F16(a1, b0[t], acc[1][t]);
    }

    // set 0 loads for next iteration (wrap harmlessly on last)
    const int kn = (k + 64 < D_) ? (k + 64) : 0;
    a0 = load_a_frag(Xh + mBase * D_ + kn, D_);
    a1 = load_a_frag(Xh + (mBase + 16) * D_ + kn, D_);
#pragma unroll
    for (int t = 0; t < 4; ++t) b0[t] = load_b_frag(Wh + (nBase + t * 16) * D_ + kn, D_);

#pragma unroll
    for (int t = 0; t < 4; ++t) {
      acc[0][t] = WMMA_F32_F16(c0, b1[t], acc[0][t]);
      acc[1][t] = WMMA_F32_F16(c1, b1[t], acc[1][t]);
    }
  }

  const int lane  = threadIdx.x & 31;
  const int col   = lane & 15;
  const int rbase = (lane >> 4) * 8;
#pragma unroll
  for (int g = 0; g < 2; ++g) {
#pragma unroll
    for (int t = 0; t < 4; ++t) {
      const int   n  = nBase + t * 16 + col;
      const float bv = bias[n];
      const int h = n >> 6, hd = n & 63;
#pragma unroll
      for (int r = 0; r < 8; ++r) {
        const int m = mBase + g * 16 + rbase + r;
        const int b = m >> 11, s = m & (S_ - 1);
        const float val = acc[g][t][r] + bv;
        long idx;
        if (mode == 2) idx = ((long)(b * H_ + h) * HD_ + hd) * S_ + s;   // v^T
        else           idx = ((long)(b * H_ + h) * S_ + s) * HD_ + hd;   // q,k
        out[idx] = (_Float16)val;
      }
    }
  }
}

// ---------------------------------------------------------------------------
// Flash attention, causal. One wave per 16-query tile.
// qh,kh: [B,H,S,64] f16; vTh: [B,H,64,S] f16; zh: [B,S,D] f16 (heads merged).
// K fragments for the next key block are prefetched one stage ahead.
// ---------------------------------------------------------------------------
__global__ __launch_bounds__(32)
void attn_kernel(const _Float16* __restrict__ qh, const _Float16* __restrict__ kh,
                 const _Float16* __restrict__ vTh, _Float16* __restrict__ zh) {
  __shared__ _Float16 lds_p[16 * 32];

  const int  lane  = threadIdx.x & 31;
  const int  qBase = blockIdx.x * 16;
  const int  h     = blockIdx.y;
  const int  b     = blockIdx.z;
  const long bh    = (long)(b * H_ + h);

  const _Float16* qp    = qh  + (bh * S_ + qBase) * HD_;
  const _Float16* kbase = kh  + bh * S_ * HD_;
  const _Float16* vbase = vTh + bh * (long)HD_ * S_;

  const v16h a0 = load_a_frag(qp, HD_);        // q features 0..31
  const v16h a1 = load_a_frag(qp + 32, HD_);   // q features 32..63

  float mi[8], li[8];
  v8f acc[4] = {};
#pragma unroll
  for (int r = 0; r < 8; ++r) { mi[r] = -3.0e38f; li[r] = 0.0f; }

  const int   col   = lane & 15;
  const int   rbase = (lane >> 4) * 8;
  const float scale = 0.125f;                  // 1/sqrt(D/H) = 1/sqrt(64)

  // one key-block stage: scores from kf, preload kfn for kb+32, softmax, P@V
  auto stage = [&](int kb, v16h (&kf)[4], v16h (&kfn)[4]) {
    v8f s0 = {}, s1 = {};
    s0 = WMMA_F32_F16(a0, kf[0], s0);
    s0 = WMMA_F32_F16(a1, kf[1], s0);
    s1 = WMMA_F32_F16(a0, kf[2], s1);
    s1 = WMMA_F32_F16(a1, kf[3], s1);

    // V fragments for this block: independent of softmax, overlap VALU below
    v16h vb[4];
#pragma unroll
    for (int t = 0; t < 4; ++t) vb[t] = load_b_frag(vbase + (t * 16) * S_ + kb, S_);

    // preload next K tile (wrap harmlessly at the end)
    const int knb = (kb + 32 < S_) ? (kb + 32) : 0;
    const _Float16* kpn = kbase + knb * HD_;
    kfn[0] = load_b_frag(kpn, HD_);
    kfn[1] = load_b_frag(kpn + 32, HD_);
    kfn[2] = load_b_frag(kpn + 16 * HD_, HD_);
    kfn[3] = load_b_frag(kpn + 16 * HD_ + 32, HD_);

#pragma unroll
    for (int r = 0; r < 8; ++r) {
      const int row = qBase + rbase + r;
      const int c0  = kb + col, c1 = c0 + 16;
      float v0 = (c0 <= row) ? s0[r] * scale : -1.0e9f;
      float v1 = (c1 <= row) ? s1[r] * scale : -1.0e9f;

      float rm = fmaxf(v0, v1);
#pragma unroll
      for (int m = 1; m < 16; m <<= 1) rm = fmaxf(rm, __shfl_xor(rm, m, 16));
      const float mn = fmaxf(mi[r], rm);
      const float f  = __expf(mi[r] - mn);
      const float p0 = __expf(v0 - mn);
      const float p1 = __expf(v1 - mn);
      float rs = p0 + p1;
#pragma unroll
      for (int m = 1; m < 16; m <<= 1) rs += __shfl_xor(rs, m, 16);
      li[r] = li[r] * f + rs;
      mi[r] = mn;
#pragma unroll
      for (int t = 0; t < 4; ++t) acc[t][r] *= f;

      const int lrow = rbase + r;               // C layout -> (row, col) in LDS
      lds_p[lrow * 32 + col]      = (_Float16)p0;
      lds_p[lrow * 32 + col + 16] = (_Float16)p1;
    }

    __asm__ volatile("s_wait_dscnt 0" ::: "memory");  // LDS stores visible wave-wide

    const v16h pA = load_a_frag(lds_p, 32);           // P as A-fragment (16x32)
#pragma unroll
    for (int t = 0; t < 4; ++t) {
      acc[t] = WMMA_F32_F16(pA, vb[t], acc[t]);       // acc += P @ V
    }
  };

  // prologue: K fragments for kb = 0
  v16h kfA[4], kfB[4];
  kfA[0] = load_b_frag(kbase, HD_);
  kfA[1] = load_b_frag(kbase + 32, HD_);
  kfA[2] = load_b_frag(kbase + 16 * HD_, HD_);
  kfA[3] = load_b_frag(kbase + 16 * HD_ + 32, HD_);

  const int end = qBase + 16;
  int kb = 0;
  while (true) {
    stage(kb, kfA, kfB);
    kb += 32; if (kb >= end) break;
    stage(kb, kfB, kfA);
    kb += 32; if (kb >= end) break;
  }

  // epilogue: z = acc / l ; write merged-head layout [B,S,D]
#pragma unroll
  for (int t = 0; t < 4; ++t) {
#pragma unroll
    for (int r = 0; r < 8; ++r) {
      const int  row = qBase + rbase + r;
      const float z  = acc[t][r] / li[r];
      const long idx = ((long)b * S_ + row) * D_ + h * HD_ + t * 16 + col;
      zh[idx] = (_Float16)z;
    }
  }
}

// ---------------------------------------------------------------------------
// Output projection: out = Z @ Wo^T + bo, f32 output. 32x64 tile per wave.
// ---------------------------------------------------------------------------
__global__ __launch_bounds__(128)
void out_proj_kernel(const _Float16* __restrict__ Zh, const _Float16* __restrict__ Wh,
                     const float* __restrict__ bias, float* __restrict__ out) {
  const int wave  = threadIdx.x >> 5;
  const int mBase = (blockIdx.x * 4 + wave) * 32;
  const int nBase = blockIdx.y * 64;

  v8f acc[2][4] = {};

  v16h a0 = load_a_frag(Zh + mBase * D_, D_);
  v16h a1 = load_a_frag(Zh + (mBase + 16) * D_, D_);
  v16h b0[4];
#pragma unroll
  for (int t = 0; t < 4; ++t) b0[t] = load_b_frag(Wh + (nBase + t * 16) * D_, D_);

#pragma unroll 1
  for (int k = 0; k < D_; k += 64) {
    v16h c0 = load_a_frag(Zh + mBase * D_ + k + 32, D_);
    v16h c1 = load_a_frag(Zh + (mBase + 16) * D_ + k + 32, D_);
    v16h b1[4];
#pragma unroll
    for (int t = 0; t < 4; ++t) b1[t] = load_b_frag(Wh + (nBase + t * 16) * D_ + k + 32, D_);

#pragma unroll
    for (int t = 0; t < 4; ++t) {
      acc[0][t] = WMMA_F32_F16(a0, b0[t], acc[0][t]);
      acc[1][t] = WMMA_F32_F16(a1, b0[t], acc[1][t]);
    }

    const int kn = (k + 64 < D_) ? (k + 64) : 0;
    a0 = load_a_frag(Zh + mBase * D_ + kn, D_);
    a1 = load_a_frag(Zh + (mBase + 16) * D_ + kn, D_);
#pragma unroll
    for (int t = 0; t < 4; ++t) b0[t] = load_b_frag(Wh + (nBase + t * 16) * D_ + kn, D_);

#pragma unroll
    for (int t = 0; t < 4; ++t) {
      acc[0][t] = WMMA_F32_F16(c0, b1[t], acc[0][t]);
      acc[1][t] = WMMA_F32_F16(c1, b1[t], acc[1][t]);
    }
  }

  const int lane  = threadIdx.x & 31;
  const int col   = lane & 15;
  const int rbase = (lane >> 4) * 8;
#pragma unroll
  for (int g = 0; g < 2; ++g) {
#pragma unroll
    for (int t = 0; t < 4; ++t) {
      const int   n  = nBase + t * 16 + col;
      const float bv = bias[n];
#pragma unroll
      for (int r = 0; r < 8; ++r) {
        const int m = mBase + g * 16 + rbase + r;
        out[(long)m * D_ + n] = acc[g][t][r] + bv;
      }
    }
  }
}

// ---------------------------------------------------------------------------
// Host launcher
// ---------------------------------------------------------------------------
extern "C" void kernel_launch(void* const* d_in, const int* in_sizes, int n_in,
                              void* d_out, int out_size, void* d_ws, size_t ws_size,
                              hipStream_t stream) {
  const float* Q  = (const float*)d_in[0];
  const float* K  = (const float*)d_in[1];
  const float* V  = (const float*)d_in[2];
  // d_in[3] = causal mask (implemented analytically)
  const float* Wq = (const float*)d_in[4];
  const float* bq = (const float*)d_in[5];
  const float* Wk = (const float*)d_in[6];
  const float* bk = (const float*)d_in[7];
  const float* Wv = (const float*)d_in[8];
  const float* bv = (const float*)d_in[9];
  const float* Wo = (const float*)d_in[10];
  const float* bo = (const float*)d_in[11];

  const size_t NX = (size_t)B_ * S_ * D_;  // 4,194,304
  const size_t NW = (size_t)D_ * D_;       // 1,048,576

  _Float16* p   = (_Float16*)d_ws;
  _Float16* Qh  = p; p += NX;
  _Float16* Kh  = p; p += NX;
  _Float16* Vh  = p; p += NX;
  _Float16* Wqh = p; p += NW;
  _Float16* Wkh = p; p += NW;
  _Float16* Wvh = p; p += NW;
  _Float16* Woh = p; p += NW;
  _Float16* qh  = p; p += NX;
  _Float16* kh  = p; p += NX;
  _Float16* vTh = p; p += NX;
  _Float16* zh  = p; p += NX;

  auto cvt = [&](const float* s, _Float16* d, size_t n) {
    int n4 = (int)(n / 4);
    f32_to_f16_kernel<<<dim3((n4 + 255) / 256), dim3(256), 0, stream>>>(s, d, n4);
  };
  cvt(Q,  Qh,  NX);
  cvt(K,  Kh,  NX);
  cvt(V,  Vh,  NX);
  cvt(Wq, Wqh, NW);
  cvt(Wk, Wkh, NW);
  cvt(Wv, Wvh, NW);
  cvt(Wo, Woh, NW);

  const int M = B_ * S_;                    // 4096
  dim3 gemmGrid(M / 128, D_ / 64);          // (32, 16): 4 waves x 32 rows each
  proj_kernel<<<gemmGrid, dim3(128), 0, stream>>>(Qh, Wqh, bq, qh,  0);
  proj_kernel<<<gemmGrid, dim3(128), 0, stream>>>(Kh, Wkh, bk, kh,  1);
  proj_kernel<<<gemmGrid, dim3(128), 0, stream>>>(Vh, Wvh, bv, vTh, 2);

  dim3 attnGrid(S_ / 16, H_, B_);           // (128, 16, 2)
  attn_kernel<<<attnGrid, dim3(32), 0, stream>>>(qh, kh, vTh, zh);

  out_proj_kernel<<<gemmGrid, dim3(128), 0, stream>>>(zh, Woh, bo, (float*)d_out);
}
